// NounPhraseModelV4_81252191306705
// MI455X (gfx1250) — matrix operand order repaired and checked by reading, not compiled, not run on hardware
//
#include <hip/hip_runtime.h>
#include <math.h>

// ---------------- problem sizes (fixed by reference) ----------------
#define NGRP  2048
#define LSLOT 64
#define QTOT  4096
#define DDIM  256
#define NTYPE 64
#define GSEG  16384
#define MMEM  131072
#define BEX   1024

typedef __attribute__((ext_vector_type(16))) __bf16 v16bf;
typedef __attribute__((ext_vector_type(8)))  float  v8f;

// ---------------- helpers ----------------
__device__ __forceinline__ int lower_bound_i32(const int* __restrict__ a, int n, int v) {
    int lo = 0, hi = n;
    while (lo < hi) { int mid = (lo + hi) >> 1; if (a[mid] < v) lo = mid + 1; else hi = mid; }
    return lo;
}

// A fragment (16x32 bf16) from row-major f32 A with leading dim lda.
// Contiguous-k layout (this code's choice): lane l holds row (l&15); 16 contiguous K elems at
// k0 + (l>>4)*16 + e.
__device__ __forceinline__ v16bf load_a_frag_f32(const float* __restrict__ A, int lda,
                                                 int rowBase, int k0, int lane) {
    const float* p = A + (rowBase + (lane & 15)) * lda + k0 + ((lane >> 4) << 4);
    v16bf a;
#pragma unroll
    for (int e = 0; e < 8; ++e) a[e] = (__bf16)p[e];
#pragma unroll
    for (int e = 0; e < 8; ++e) a[8 + e] = (__bf16)p[8 + e];
    return a;
}

// B fragment (32x16 bf16) where B = W^T, W row-major [N][K]:
// B(k,n) = W[n][k]. Lane l holds col n=(l&15); 16 contiguous K elems at
// k0 + (l>>4)*16.
__device__ __forceinline__ v16bf load_b_frag_wt(const float* __restrict__ W, int ldw,
                                                int colBase, int k0, int lane) {
    const float* p = W + (colBase + (lane & 15)) * ldw + k0 + ((lane >> 4) << 4);
    v16bf b;
#pragma unroll
    for (int e = 0; e < 16; ++e) b[e] = (__bf16)p[e];
    return b;
}

// ---------------- K1: x_sum = segment_sum(emb[mem], grp) ----------------
// grp is sorted; one block per segment, binary-search the range. Deterministic.
__global__ __launch_bounds__(256) void seg_sum_kernel(const float* __restrict__ emb,
                                                      const int* __restrict__ mem,
                                                      const int* __restrict__ grp,
                                                      float* __restrict__ xsum) {
    __shared__ int sh[2];
    int g = blockIdx.x;
    if (threadIdx.x == 0) {
        sh[0] = lower_bound_i32(grp, MMEM, g);
        sh[1] = lower_bound_i32(grp, MMEM, g + 1);
    }
    __syncthreads();
    int lo = sh[0], hi = sh[1];
    int t = threadIdx.x;
    float acc = 0.f;
    for (int j = lo; j < hi; ++j) {
        int tok = mem[j];
        acc += emb[(size_t)tok * DDIM + t];
    }
    xsum[(size_t)g * DDIM + t] = acc;
}

// ---------------- K2: C[M,256] = A[M,256] * W[256,256]^T + bias ------------
// bf16 WMMA with f32 accumulate. One wave per 16x16 output tile.
__global__ __launch_bounds__(256) void gemm_bf16_nt(const float* __restrict__ A,
                                                    const float* __restrict__ W,
                                                    const float* __restrict__ bias,
                                                    float* __restrict__ C, int Mrows) {
    const int K = 256, N = 256;
    int wave = (blockIdx.x * blockDim.x + threadIdx.x) >> 5;
    int lane = threadIdx.x & 31;
    int mt = wave >> 4;        // N/16 == 16 tiles along N
    int nt = wave & 15;
    if (mt * 16 >= Mrows) return;
    v8f acc = {};
#pragma unroll
    for (int k0 = 0; k0 < K; k0 += 32) {
        v16bf a = load_a_frag_f32(A, K, mt * 16, k0, lane);
        v16bf b = load_b_frag_wt(W, K, nt * 16, k0, lane);
        acc = __builtin_amdgcn_wmma_f32_16x16x32_bf16(false, a, false, b, (short)0, acc,
                                                      false, false);
    }
    int col = nt * 16 + (lane & 15);
    float bv = bias[col];
    int rbase = mt * 16 + ((lane >> 4) << 3);
#pragma unroll
    for (int r = 0; r < 8; ++r) C[(size_t)(rbase + r) * N + col] = acc[r] + bv;
}

// ---------------- K3: fused per-group attention ----------------
// Block = 1 group (2 queries, 64 kv slots), 256 threads = 8 waves.
// Phase 1: kz = k[g] @ Wk^T + bk via 512 bf16 WMMA ops -> LDS (bf16).
// Phase 2-5: scores, masked softmax, att = p @ x_sum[pos], relation logits,
// per-group metric partials (fixed-order sums; no float atomics).
#define LDK 258  // padded bf16 row stride: (258*2)/4 % 64 == 1 -> conflict-free columns
__global__ __launch_bounds__(256) void attn_kernel(
    const float* __restrict__ q, const float* __restrict__ k,
    const float* __restrict__ Wk, const float* __restrict__ bk,
    const float* __restrict__ Wrel, const float* __restrict__ brel,
    const unsigned char* __restrict__ m, const int* __restrict__ pos2grp,
    const unsigned char* __restrict__ typ, const float* __restrict__ qz,
    const float* __restrict__ xsum, float* __restrict__ logit_out,
    float* __restrict__ nll_part, int* __restrict__ correct_cnt,
    int* __restrict__ eqall) {
    __shared__ __bf16 kzl[LSLOT * LDK];   // 33 KB
    __shared__ float qlds[2][DDIM];
    __shared__ float qzlds[2][DDIM];
    __shared__ float attl[2][DDIM];
    __shared__ float slds[2][LSLOT];
    __shared__ float plds[2][LSLOT];
    __shared__ float nlll[128];
    __shared__ int   eql[128];
    __shared__ int   posl[LSLOT];

    const int g = blockIdx.x;
    const int t = threadIdx.x;
    const int wave = t >> 5, lane = t & 31;

    // preload q and qz rows for the 2 queries of this group
    for (int i = t; i < 2 * DDIM; i += 256) {
        int qi = i >> 8, c = i & 255;
        int qrow = 2 * g + qi;
        qlds[qi][c]  = q[(size_t)qrow * DDIM + c];
        qzlds[qi][c] = qz[(size_t)qrow * DDIM + c];
    }
    if (t < LSLOT) posl[t] = pos2grp[g * LSLOT + t];

    // ---- Phase 1: kz (64x256) = k[g] @ Wk^T + bk, bf16 WMMA -> LDS ----
    const float* kg = k + (size_t)g * LSLOT * DDIM;
    for (int i = 0; i < 8; ++i) {
        int tile = wave + (i << 3);          // 64 tiles over 8 waves
        int mt = tile >> 4, nt = tile & 15;
        v8f acc = {};
#pragma unroll
        for (int k0 = 0; k0 < DDIM; k0 += 32) {
            v16bf a = load_a_frag_f32(kg, DDIM, mt * 16, k0, lane);
            v16bf b = load_b_frag_wt(Wk, DDIM, nt * 16, k0, lane);
            acc = __builtin_amdgcn_wmma_f32_16x16x32_bf16(false, a, false, b, (short)0,
                                                          acc, false, false);
        }
        int col = nt * 16 + (lane & 15);
        float bv = bk[col];
        int rbase = mt * 16 + ((lane >> 4) << 3);
#pragma unroll
        for (int r = 0; r < 8; ++r) kzl[(rbase + r) * LDK + col] = (__bf16)(acc[r] + bv);
    }
    __syncthreads();

    // ---- Phase 2: scores s[qi][l] = qz . kz[l] / sqrt(256), masked ----
    if (t < 128) {
        int qi = t >> 6, l = t & 63;
        float s;
        if (m[g * LSLOT + l]) {
            const __bf16* kr = &kzl[l * LDK];
            float acc = 0.f;
#pragma unroll 8
            for (int c = 0; c < DDIM; ++c) acc += qzlds[qi][c] * (float)kr[c];
            s = acc * 0.0625f;               // 1/sqrt(256)
        } else {
            s = -INFINITY;
        }
        slds[qi][l] = s;
    }
    __syncthreads();

    // ---- Phase 3: softmax over l (tiny, serial per query) ----
    if (t < 2) {
        float mx = -INFINITY;
        for (int l = 0; l < LSLOT; ++l) mx = fmaxf(mx, slds[t][l]);
        float sum = 0.f;
        for (int l = 0; l < LSLOT; ++l) {
            float e = expf(slds[t][l] - mx);  // exp(-inf - mx) == 0
            plds[t][l] = e;
            sum += e;
        }
        float inv = 1.f / sum;
        for (int l = 0; l < LSLOT; ++l) plds[t][l] *= inv;
    }
    __syncthreads();

    // ---- Phase 4: att[qi][dx] = sum_l p[l] * x_sum[pos[l]][dx] ----
    {
        float a0 = 0.f, a1 = 0.f;
        for (int l = 0; l < LSLOT; ++l) {
            float xv = xsum[(size_t)posl[l] * DDIM + t];
            a0 += plds[0][l] * xv;
            a1 += plds[1][l] * xv;
        }
        attl[0][t] = a0;
        attl[1][t] = a1;
    }
    __syncthreads();

    // ---- Phase 5: logit = [q | att] @ Wrel^T + brel ; metrics ----
    if (t < 128) {
        int qi = t >> 6, j = t & 63;
        int qrow = 2 * g + qi;
        const float* wr = Wrel + (size_t)j * (2 * DDIM);
        float acc = brel[j];
        for (int c = 0; c < DDIM; ++c) acc += qlds[qi][c] * wr[c];
        for (int c = 0; c < DDIM; ++c) acc += attl[qi][c] * wr[DDIM + c];
        logit_out[(size_t)qrow * NTYPE + j] = acc;

        bool gt = acc > 0.f;
        bool tv = typ[(size_t)qrow * NTYPE + j] != 0;
        eql[t] = (gt == tv) ? 1 : 0;
        // stable binary_cross_entropy_with_logits element
        nlll[t] = fmaxf(acc, 0.f) + log1pf(expf(-fabsf(acc))) - acc * (tv ? 1.f : 0.f);
    }
    __syncthreads();

    if (t < 2) {                     // per-query count + all-reduce, fixed order
        int cnt = 0, allv = 1;
        for (int j = 0; j < NTYPE; ++j) {
            int e = eql[t * 64 + j];
            cnt += e;
            allv &= e;
        }
        correct_cnt[2 * g + t] = cnt;
        eqall[2 * g + t] = allv;
    }
    if (t == 0) {                    // group nll partial, fixed order
        float s = 0.f;
        for (int j = 0; j < 128; ++j) s += nlll[j];
        nll_part[g] = s;
    }
}

// ---------------- K4a: em[b] = segment_min(eqall, idx) (idx sorted) --------
__global__ __launch_bounds__(256) void em_kernel(const int* __restrict__ idx,
                                                 const int* __restrict__ eqall,
                                                 float* __restrict__ em_out) {
    int b = blockIdx.x * blockDim.x + threadIdx.x;
    if (b >= BEX) return;
    int lo = lower_bound_i32(idx, QTOT, b);
    int hi = lower_bound_i32(idx, QTOT, b + 1);
    int mn = 2147483647;             // jax segment_min identity for empty segments
    for (int j = lo; j < hi; ++j) mn = min(mn, eqall[j]);
    em_out[b] = (float)mn;
}

// ---------------- K4b: deterministic scalar reductions ----------------
__global__ __launch_bounds__(256) void finalize_kernel(const int* __restrict__ correct,
                                                       const float* __restrict__ nllp,
                                                       const float* __restrict__ em,
                                                       float* __restrict__ outAcc,
                                                       float* __restrict__ outEmr,
                                                       float* __restrict__ outNll) {
    __shared__ float sf[256];
    __shared__ int   si[256];
    int t = threadIdx.x;

    int ci = 0;
    for (int j = t; j < QTOT; j += 256) ci += correct[j];
    si[t] = ci;
    __syncthreads();
    if (t == 0) {
        long s = 0;
        for (int i = 0; i < 256; ++i) s += si[i];
        *outAcc = (float)((double)s / (double)(QTOT * (long)NTYPE));
    }
    __syncthreads();

    float nf = 0.f;
    for (int j = t; j < NGRP; j += 256) nf += nllp[j];
    sf[t] = nf;
    __syncthreads();
    if (t == 0) {
        float s = 0.f;
        for (int i = 0; i < 256; ++i) s += sf[i];
        *outNll = s / (float)NGRP;
    }
    __syncthreads();

    float ef = 0.f;
    for (int j = t; j < BEX; j += 256) ef += em[j];
    sf[t] = ef;
    __syncthreads();
    if (t == 0) {
        float s = 0.f;
        for (int i = 0; i < 256; ++i) s += sf[i];
        *outEmr = s / (float)BEX;
    }
}

// ---------------- launch ----------------
extern "C" void kernel_launch(void* const* d_in, const int* in_sizes, int n_in,
                              void* d_out, int out_size, void* d_ws, size_t ws_size,
                              hipStream_t stream) {
    (void)in_sizes; (void)n_in; (void)out_size; (void)ws_size;
    const float* q    = (const float*)d_in[0];
    const float* k    = (const float*)d_in[1];
    const float* emb  = (const float*)d_in[2];
    const float* Wq   = (const float*)d_in[3];
    const float* bq   = (const float*)d_in[4];
    const float* Wk   = (const float*)d_in[5];
    const float* bk   = (const float*)d_in[6];
    const float* Wrel = (const float*)d_in[7];
    const float* brel = (const float*)d_in[8];
    const unsigned char* m   = (const unsigned char*)d_in[9];
    /* d_in[10] = qgrp: implied by 2 queries/group, unused */
    const int* mem     = (const int*)d_in[11];
    const int* grp     = (const int*)d_in[12];
    const int* pos2grp = (const int*)d_in[13];
    const unsigned char* typ = (const unsigned char*)d_in[14];
    const int* idx     = (const int*)d_in[15];

    float* out       = (float*)d_out;
    float* logit_out = out;                         // Q*NTYPE = 262144
    float* accp      = out + (size_t)QTOT * NTYPE;  // +0
    float* emrp      = accp + 1;
    float* nllo      = accp + 2;
    float* em_out    = accp + 3;                    // 1024

    char* ws = (char*)d_ws;
    float* xsum     = (float*)ws;                         // GSEG*256 f32
    float* qz       = xsum + (size_t)GSEG * DDIM;         // QTOT*256 f32
    float* nll_part = qz + (size_t)QTOT * DDIM;           // NGRP f32
    int*   correct  = (int*)(nll_part + NGRP);            // QTOT i32
    int*   eqall    = correct + QTOT;                     // QTOT i32

    // 1) segment-sum of token embeddings (sorted grp -> deterministic)
    seg_sum_kernel<<<GSEG, 256, 0, stream>>>(emb, mem, grp, xsum);
    // 2) qz = q @ Wq^T + bq  (bf16 WMMA; 4096 waves = 512 blocks)
    gemm_bf16_nt<<<(QTOT / 16) * 16 / 8, 256, 0, stream>>>(q, Wq, bq, qz, QTOT);
    // 3) fused kz-GEMM + masked softmax attention + relation logits + metrics
    attn_kernel<<<NGRP, 256, 0, stream>>>(q, k, Wk, bk, Wrel, brel, m, pos2grp, typ,
                                          qz, xsum, logit_out, nll_part, correct, eqall);
    // 4) exact-match segment-min + scalar reductions
    em_kernel<<<BEX / 256, 256, 0, stream>>>(idx, eqall, em_out);
    finalize_kernel<<<1, 256, 0, stream>>>(correct, nll_part, em_out, accp, emrp, nllo);
}
